// ActionDiffusionModel_64647847739636
// MI455X (gfx1250) — matrix operand order repaired and checked by reading, not compiled, not run on hardware
//
#include <hip/hip_runtime.h>

typedef __attribute__((ext_vector_type(16))) _Float16 v16h;
typedef __attribute__((ext_vector_type(8)))  _Float16 h8;
typedef __attribute__((ext_vector_type(4)))  _Float16 h4;
typedef __attribute__((ext_vector_type(8)))  float    v8f;
typedef __attribute__((ext_vector_type(4)))  float    f4;
typedef __attribute__((ext_vector_type(4)))  int      i4;

#define GLOBAL_AS __attribute__((address_space(1)))
#define LDS_AS    __attribute__((address_space(3)))

#if __has_builtin(__builtin_amdgcn_global_load_async_to_lds_b128)
#define HAS_ASYNC_LDS 1
#endif

// 16-byte global -> LDS copy: async (ASYNCcnt-tracked, no VGPR dest) if the
// gfx1250 builtin exists, else a plain synchronous copy through VGPRs.
__device__ __forceinline__ void async_cp16(const void* g, void* l) {
#ifdef HAS_ASYNC_LDS
  __builtin_amdgcn_global_load_async_to_lds_b128(
      (GLOBAL_AS i4*)g, (LDS_AS i4*)l, 0, 0);
#else
  *(f4*)l = *(const f4*)g;
#endif
}

#ifdef HAS_ASYNC_LDS
#if __has_builtin(__builtin_amdgcn_s_wait_asynccnt)
#define WAIT_ASYNC(n) __builtin_amdgcn_s_wait_asynccnt(n)
#else
#define WAIT_ASYNC(n) asm volatile("s_wait_asynccnt %0" ::"n"(n) : "memory")
#endif
#else
#define WAIT_ASYNC(n)
#endif

#define C_DIM   1024
#define NSLAB   16              // C_DIM / 64 (two WMMA k-steps per slab)
#define MCHUNK  1024            // bank rows per workgroup
#define NITER   8               // MCHUNK / 128
#define KSEL    16
#define SCP     129             // score stride (conflict-free: 129 % 64 == 1)
#define BSTRIDE 68              // bank slab row stride (f32): 64 data + 4 pad
#define QSTRIDE 72              // query slab row stride (f16): 64 data + 8 pad
#define BSTG    (128 * BSTRIDE) // floats per bank slab buffer (34 KB)
#define QSTG    (256 * QSTRIDE) // f16 per query slab buffer (36 KB)
#define QBASE   (2 * BSTG)      // float offset of query staging region
#define SMEMF   (QBASE + QSTG)  // 35840 floats = 140 KB total (scores alias)

// ---------------- Kernel 0: query f32 -> f16 (one-time, 1 MB) ----------------
__global__ __launch_bounds__(256) void cvt_query(const float* __restrict__ q,
                                                 _Float16* __restrict__ qf) {
  int i = (blockIdx.x * 256 + threadIdx.x) * 4;
  f4 x = *(const f4*)(q + i);
  h4 y;
  y[0] = (_Float16)x.x; y[1] = (_Float16)x.y;
  y[2] = (_Float16)x.z; y[3] = (_Float16)x.w;
  *(h4*)(qf + i) = y;
}

// Stage one 64-wide K slab (bank 128x64 f32 + query 256x64 f16 = 64 KB) into
// LDS buffer `buf` via 16 async b128 ops per lane (uniform across all waves).
__device__ __forceinline__ void stage_slab(
    const float* __restrict__ bank, const _Float16* __restrict__ qf,
    float* __restrict__ smem, int buf, long mBase, int slab,
    int w, int l, int M) {
  float* bankS = smem + buf * BSTG;
  _Float16* qS = (_Float16*)(smem + QBASE) + buf * QSTG;
#pragma unroll
  for (int i = 0; i < 8; ++i) {          // bank: 2048 x 16B units over 256 lanes
    int U = w * 256 + i * 32 + l;
    int row = U >> 4, part = U & 15;     // row 0..127, 16B chunk 0..15
    long rg = mBase + row;
    if (rg >= M) rg = 0;                 // OOB rows filtered by index later
    async_cp16(bank + rg * (long)C_DIM + slab * 64 + part * 4,
               bankS + row * BSTRIDE + part * 4);
  }
#pragma unroll
  for (int i = 0; i < 8; ++i) {          // query: 2048 x 16B units
    int V = w * 256 + i * 32 + l;
    int qrow = V >> 3, qp = V & 7;       // qrow 0..255, 16B chunk 0..7
    async_cp16(qf + (long)qrow * C_DIM + slab * 64 + qp * 8,
               qS + qrow * QSTRIDE + qp * 8);
  }
}

// ------- Kernel 1: async-staged WMMA distance GEMM + per-chunk top-16 -------
__global__ __launch_bounds__(256) void knn_part(
    const _Float16* __restrict__ qf, const float* __restrict__ bank,
    float* __restrict__ part_s, int* __restrict__ part_i, int M) {
  __shared__ float smem[SMEMF];          // 140 KB; scores (129 KB) alias staging

  const int t    = threadIdx.x;
  const int w    = t >> 5;
  const int l    = t & 31;
  const int ln   = l & 15;
  const int half = l >> 4;
  const int qg   = w & 3;                // 64-query group
  const int ng   = w >> 2;               // 64-bank-row group
  const long chunkBase = (long)blockIdx.x * MCHUNK;

  float ts[KSEL]; int ti[KSEL];
#pragma unroll
  for (int j = 0; j < KSEL; ++j) { ts[j] = 3.4e38f; ti[j] = 0; }

  for (int it = 0; it < NITER; ++it) {
    const long mBase = chunkBase + (long)it * 128;

    v8f acc[4][4];
    v8f zero = {};
#pragma unroll
    for (int a = 0; a < 4; ++a)
#pragma unroll
      for (int b = 0; b < 4; ++b) acc[a][b] = zero;
    float psq[4] = {0.f, 0.f, 0.f, 0.f};

    stage_slab(bank, qf, smem, 0, mBase, 0, w, l, M);   // prologue: slab 0

    for (int s = 0; s < NSLAB; ++s) {
      const int bufc = s & 1;
      WAIT_ASYNC(0);           // drain own slab-s async ops
      __syncthreads();         // all waves drained -> slab s published,
                               // and old buffer is free for reuse
      if (s + 1 < NSLAB)       // stage next slab while we compute this one
        stage_slab(bank, qf, smem, bufc ^ 1, mBase, s + 1, w, l, M);

      const float* bb = smem + bufc * BSTG;
      const _Float16* qb = (const _Float16*)(smem + QBASE) + bufc * QSTG;
#pragma unroll
      for (int kk = 0; kk < 2; ++kk) {   // two WMMA k-steps per slab
        // ---- issue ALL 24 ds loads for this k-step first (one clause) ----
        f4 braw[4][4];
#pragma unroll
        for (int nt = 0; nt < 4; ++nt) {
          const float* p = bb + (ng * 64 + nt * 16 + ln) * BSTRIDE
                           + kk * 32 + half * 16;
#pragma unroll
          for (int c = 0; c < 4; ++c) braw[nt][c] = *(const f4*)(p + c * 4);
        }
        h8 alo[4], ahi[4];
#pragma unroll
        for (int tq = 0; tq < 4; ++tq) {
          const _Float16* qp = qb + ((qg * 4 + tq) * 16 + ln) * QSTRIDE
                               + kk * 32 + half * 16;
          alo[tq] = *(const h8*)(qp);
          ahi[tq] = *(const h8*)(qp + 8);
        }
        // ---- convert B (f32 -> f16) + fold ||p||^2 ----
        v16h bfrag[4];
#pragma unroll
        for (int nt = 0; nt < 4; ++nt) {
          v16h bf;
#pragma unroll
          for (int c = 0; c < 4; ++c) {
            f4 x = braw[nt][c];
            psq[nt] += x.x * x.x + x.y * x.y + x.z * x.z + x.w * x.w;
            bf[c * 4 + 0] = (_Float16)x.x;
            bf[c * 4 + 1] = (_Float16)x.y;
            bf[c * 4 + 2] = (_Float16)x.z;
            bf[c * 4 + 3] = (_Float16)x.w;
          }
          bfrag[nt] = bf;
        }
        // ---- 16 WMMAs ----
#pragma unroll
        for (int tq = 0; tq < 4; ++tq) {
          v16h af = __builtin_shufflevector(alo[tq], ahi[tq],
                                            0, 1, 2, 3, 4, 5, 6, 7,
                                            8, 9, 10, 11, 12, 13, 14, 15);
#pragma unroll
          for (int nt = 0; nt < 4; ++nt)
            acc[tq][nt] = __builtin_amdgcn_wmma_f32_16x16x32_f16(
                false, af, false, bfrag[nt], (short)0, acc[tq][nt], false, false);
        }
      }
    }
    __syncthreads();   // all staging reads done before scores overwrite it

    // score = ||p||^2 - 2 q.p  ->  LDS [query][localN]
#pragma unroll
    for (int nt = 0; nt < 4; ++nt) {
      float pf = psq[nt] + __shfl_xor(psq[nt], 16, 32);   // join K halves
      int nl = ng * 64 + nt * 16 + ln;
#pragma unroll
      for (int tq = 0; tq < 4; ++tq)
#pragma unroll
        for (int v = 0; v < 8; ++v) {
          int q = (qg * 4 + tq) * 16 + half * 8 + v;      // C-layout: M=v+8*half
          smem[q * SCP + nl] = pf - 2.0f * acc[tq][nt][v];
        }
    }
    __syncthreads();

    // thread t == query t: fold 128 candidates into sorted top-16 registers
#pragma unroll 4
    for (int j = 0; j < 128; ++j) {
      float s = smem[t * SCP + j];
      long gi = mBase + j;
      if (gi < M && s < ts[KSEL - 1]) {
        ts[KSEL - 1] = s; ti[KSEL - 1] = (int)gi;
#pragma unroll
        for (int jj = KSEL - 1; jj > 0; --jj) {
          if (ts[jj] < ts[jj - 1]) {
            float tf = ts[jj]; ts[jj] = ts[jj - 1]; ts[jj - 1] = tf;
            int tg = ti[jj]; ti[jj] = ti[jj - 1]; ti[jj - 1] = tg;
          }
        }
      }
    }
    __syncthreads();   // selection done before next iter's staging writes
  }

#pragma unroll
  for (int j = 0; j < KSEL; ++j) {
    size_t o = ((size_t)blockIdx.x * 256 + t) * KSEL + j;
    part_s[o] = ts[j];
    part_i[o] = ti[j];
  }
}

// ------- Kernel 2: merge partial top-16 lists + gather trajectories ---------
__global__ __launch_bounds__(256) void knn_merge_gather(
    const float* __restrict__ part_s, const int* __restrict__ part_i,
    const float* __restrict__ traj, float* __restrict__ out, int ncand) {
  __shared__ float cs[4096];
  __shared__ int   ci[4096];
  __shared__ float rs[256];
  __shared__ int   rp[256];
  __shared__ int   sel[KSEL];

  const int b = blockIdx.x, t = threadIdx.x;
  for (int j = t; j < ncand; j += 256) {
    int wg = j >> 4, jj = j & 15;
    cs[j] = part_s[((size_t)wg * 256 + b) * KSEL + jj];
    ci[j] = part_i[((size_t)wg * 256 + b) * KSEL + jj];
  }
  __syncthreads();

  for (int r = 0; r < KSEL; ++r) {
    float best = 3.4e38f; int bp = 0;
    for (int j = t; j < ncand; j += 256)
      if (cs[j] < best) { best = cs[j]; bp = j; }
    rs[t] = best; rp[t] = bp;
    __syncthreads();
    for (int off = 128; off > 0; off >>= 1) {
      if (t < off && rs[t + off] < rs[t]) { rs[t] = rs[t + off]; rp[t] = rp[t + off]; }
      __syncthreads();
    }
    if (t == 0) { sel[r] = ci[rp[0]]; cs[rp[0]] = 3.4e38f; }
    __syncthreads();
  }

  for (int e = t; e < KSEL * 24; e += 256) {   // out[b][k][8][3]
    int kk = e / 24, rr = e % 24;
    out[(size_t)b * (KSEL * 24) + e] = traj[(size_t)sel[kk] * 24 + rr];
  }
}

extern "C" void kernel_launch(void* const* d_in, const int* in_sizes, int n_in,
                              void* d_out, int out_size, void* d_ws, size_t ws_size,
                              hipStream_t stream) {
  const float* query = (const float*)d_in[0];   // [256,1024] f32
  const float* bank  = (const float*)d_in[1];   // [M,1024]  f32
  const float* traj  = (const float*)d_in[2];   // [M,8,3]   f32
  // d_in[3] = k (==16, baked into KSEL)

  const int M   = in_sizes[1] / C_DIM;                  // 200000
  const int NWG = (M + MCHUNK - 1) / MCHUNK;            // 196

  float*    part_s = (float*)d_ws;
  int*      part_i = (int*)(part_s + (size_t)NWG * 256 * KSEL);
  _Float16* qf16   = (_Float16*)(part_i + (size_t)NWG * 256 * KSEL);

  cvt_query<<<256, 256, 0, stream>>>(query, qf16);
  knn_part<<<NWG, 256, 0, stream>>>(qf16, bank, part_s, part_i, M);
  knn_merge_gather<<<256, 256, 0, stream>>>(part_s, part_i, traj,
                                            (float*)d_out, NWG * KSEL);
}
